// MultiHeadAttention_11785390260373
// MI455X (gfx1250) — matrix operand order repaired and checked by reading, not compiled, not run on hardware
//
#include <hip/hip_runtime.h>
#include <hip/hip_bf16.h>

#define N_EMBD 768
#define N_HEAD 12
#define HEAD   64
#define BATCH  2
#define SEQ    4096

typedef __attribute__((ext_vector_type(16))) __bf16 v16bf;
typedef __attribute__((ext_vector_type(8)))  __bf16 bf16x8;
typedef __attribute__((ext_vector_type(8)))  float  v8f;

__device__ __forceinline__ v16bf combine8(bf16x8 lo, bf16x8 hi) {
  v16bf f;
#pragma unroll
  for (int i = 0; i < 8; ++i) { f[i] = lo[i]; f[i + 8] = hi[i]; }
  return f;
}

// A-matrix fragment (16x32 bf16): per lane two contiguous 8-elem chunks at c0 and c0+16
__device__ __forceinline__ v16bf load_a_frag(const __bf16* rowp, int c0) {
  return combine8(*(const bf16x8*)(rowp + c0), *(const bf16x8*)(rowp + c0 + 16));
}

// B-matrix fragment (32x16 bf16): per lane two contiguous 8-elem chunks at p and p+16
__device__ __forceinline__ v16bf load_b_frag(const __bf16* p) {
  return combine8(*(const bf16x8*)(p), *(const bf16x8*)(p + 16));
}

__device__ __forceinline__ v8f wmma_bf16(v16bf a, v16bf b, v8f c) {
  return __builtin_amdgcn_wmma_f32_16x16x32_bf16(false, a, false, b, (short)0, c,
                                                 false, false);
}

// ---------------------------------------------------------------- cast kernel
__global__ void cast_f32_to_bf16(const float* __restrict__ in,
                                 __bf16* __restrict__ out, int n) {
  int i = blockIdx.x * blockDim.x + threadIdx.x;
  int stride = gridDim.x * blockDim.x;
  for (; i < n; i += stride) out[i] = (__bf16)in[i];
}

// ---------------------------------------------------------------- WMMA GEMM
// C[row, o] = sum_c A[row, c] * W[o, c]; each wave computes a 16(M) x 64(N) tile
// (one head-wide group), reusing one A fragment across 4 B fragments per K-step.
// mode 0: out_bf = [B, H, T, 64]     (Q / K)
// mode 1: out_bf = [B, H, 64, T]     (V transposed)
// mode 2: out_f  = [B*T, 768] fp32, + bias
__global__ __launch_bounds__(256)
void gemm_wmma(const __bf16* __restrict__ A, const __bf16* __restrict__ W,
               __bf16* __restrict__ out_bf, float* __restrict__ out_f,
               const float* __restrict__ bias, int mode) {
  const int NT64 = N_EMBD / 64;               // 12 head-wide column groups
  int wid  = threadIdx.x >> 5;
  int lane = threadIdx.x & 31;
  int tile = blockIdx.x * (blockDim.x >> 5) + wid;
  int ht = tile % NT64, mt = tile / NT64;
  int n  = lane & 15;
  int hi = lane >> 4;
  int c0a = hi ? 8 : 0;
  int c0b = hi ? 8 : 0;

  const __bf16* arow = A + (size_t)(mt * 16 + n) * N_EMBD;
  const __bf16* wr0  = W + (size_t)(ht * 64 +  0 + n) * N_EMBD;
  const __bf16* wr1  = W + (size_t)(ht * 64 + 16 + n) * N_EMBD;
  const __bf16* wr2  = W + (size_t)(ht * 64 + 32 + n) * N_EMBD;
  const __bf16* wr3  = W + (size_t)(ht * 64 + 48 + n) * N_EMBD;

  v8f acc0 = {}, acc1 = {}, acc2 = {}, acc3 = {};
#pragma unroll 2
  for (int kk = 0; kk < N_EMBD; kk += 32) {
    v16bf a = load_a_frag(arow, kk + c0a);
    acc0 = wmma_bf16(a, load_b_frag(wr0 + kk + c0b), acc0);
    acc1 = wmma_bf16(a, load_b_frag(wr1 + kk + c0b), acc1);
    acc2 = wmma_bf16(a, load_b_frag(wr2 + kk + c0b), acc2);
    acc3 = wmma_bf16(a, load_b_frag(wr3 + kk + c0b), acc3);
  }

  int mrow0 = mt * 16 + (hi ? 8 : 0);
  v8f accs[4] = {acc0, acc1, acc2, acc3};
  if (mode == 2) {
#pragma unroll
    for (int sub = 0; sub < 4; ++sub) {
      int o = ht * 64 + sub * 16 + n;
      float bv = bias[o];
#pragma unroll
      for (int r = 0; r < 8; ++r)
        out_f[(size_t)(mrow0 + r) * N_EMBD + o] = accs[sub][r] + bv;
    }
  } else {
#pragma unroll
    for (int sub = 0; sub < 4; ++sub) {
      int d = sub * 16 + n;
#pragma unroll
      for (int r = 0; r < 8; ++r) {
        int row = mrow0 + r;
        int bb = row / SEQ, t = row % SEQ;
        size_t dst;
        if (mode == 0)
          dst = (((size_t)(bb * N_HEAD + ht) * SEQ) + t) * HEAD + d;
        else
          dst = (((size_t)(bb * N_HEAD + ht) * HEAD) + d) * SEQ + t;
        out_bf[dst] = (__bf16)accs[sub][r];
      }
    }
  }
}

// -------------------------------------------------- online softmax for one S tile
__device__ __forceinline__ void softmax_update(
    v8f& s0, v8f& s1, int qtop, int kb, int n, int mrow,
    float* rowmax, float* rowsum,
    v8f& o0, v8f& o1, v8f& o2, v8f& o3, __bf16* lds) {
  const float scale = 0.125f;                 // 1/sqrt(64)
  const float NEGI  = -1e30f;
#pragma unroll
  for (int r = 0; r < 8; ++r) {
    int qi = qtop + r + mrow;
    s0[r] = (kb + n      <= qi) ? s0[r] * scale : NEGI;
    s1[r] = (kb + 16 + n <= qi) ? s1[r] * scale : NEGI;
  }
#pragma unroll
  for (int r = 0; r < 8; ++r) {
    float tm = fmaxf(s0[r], s1[r]);
#pragma unroll
    for (int off = 8; off > 0; off >>= 1)
      tm = fmaxf(tm, __shfl_xor(tm, off, 32));
    float mnew  = fmaxf(rowmax[r], tm);
    float alpha = __expf(rowmax[r] - mnew);
    rowmax[r] = mnew;
    float p0 = __expf(s0[r] - mnew);
    float p1 = __expf(s1[r] - mnew);
    float ts = p0 + p1;
#pragma unroll
    for (int off = 8; off > 0; off >>= 1)
      ts += __shfl_xor(ts, off, 32);
    rowsum[r] = rowsum[r] * alpha + ts;
    o0[r] *= alpha; o1[r] *= alpha; o2[r] *= alpha; o3[r] *= alpha;
    int m = r + mrow;
    lds[m * 32 + n]      = (__bf16)p0;
    lds[m * 32 + 16 + n] = (__bf16)p1;
  }
}

// ---------------------------------------------------------------- flash attention
// One wave = 32 queries (two 16-row tiles) of one (b, h): K/V fragments are
// loaded once per 32-key step and feed both Q tiles (16 WMMAs per step).
// Q,K: [B,H,T,64] bf16; Vt: [B,H,64,T] bf16; Y: [B*T, 768] bf16.
__global__ __launch_bounds__(256)
void attn_flash(const __bf16* __restrict__ Q, const __bf16* __restrict__ K,
                const __bf16* __restrict__ Vt, __bf16* __restrict__ Y) {
  __shared__ __align__(16) __bf16 Ptile[8][2 * 16 * 32];

  int wid  = threadIdx.x >> 5;
  int lane = threadIdx.x & 31;
  int g    = blockIdx.x * (blockDim.x >> 5) + wid;   // 32-query supertile id
  const int QT2 = SEQ / 32;                          // 128
  int qb2 = (g % QT2) * 32;
  int bh  = g / QT2;                                 // b * N_HEAD + h
  int n   = lane & 15;
  int hi  = lane >> 4;
  int mrow = hi ? 8 : 0;
  int c0a = hi ? 8 : 0;
  int c0b = hi ? 8 : 0;

  const __bf16* Qh = Q  + (size_t)bh * SEQ * HEAD;
  const __bf16* Kh = K  + (size_t)bh * SEQ * HEAD;
  const __bf16* Vh = Vt + (size_t)bh * HEAD * SEQ;

  // Q A-fragments: two q-tiles x (d 0..31, d 32..63)
  const __bf16* qrow0 = Qh + (size_t)(qb2 + n) * HEAD;
  const __bf16* qrow1 = Qh + (size_t)(qb2 + 16 + n) * HEAD;
  v16bf qa00 = load_a_frag(qrow0, 0  + c0a);
  v16bf qa01 = load_a_frag(qrow0, 32 + c0a);
  v16bf qa10 = load_a_frag(qrow1, 0  + c0a);
  v16bf qa11 = load_a_frag(qrow1, 32 + c0a);

  v8f oA[4] = {}, oB[4] = {};
  float rmA[8], rsA[8], rmB[8], rsB[8];
#pragma unroll
  for (int r = 0; r < 8; ++r) {
    rmA[r] = -1e30f; rsA[r] = 0.f;
    rmB[r] = -1e30f; rsB[r] = 0.f;
  }

  __bf16* lds  = &Ptile[wid][0];
  __bf16* lds2 = lds + 512;
  int qlast = qb2 + 31;

  for (int kb = 0; kb <= qlast; kb += 32) {
    // ---- S = Q K^T for 32 keys, both q-tiles (K frags shared)
    const __bf16* krow0 = Kh + (size_t)(kb + n) * HEAD;
    const __bf16* krow1 = Kh + (size_t)(kb + 16 + n) * HEAD;
    v16bf kf00 = load_b_frag(krow0 + 0  + c0b);
    v16bf kf01 = load_b_frag(krow0 + 32 + c0b);
    v16bf kf10 = load_b_frag(krow1 + 0  + c0b);
    v16bf kf11 = load_b_frag(krow1 + 32 + c0b);
    if (kb + 32 <= qlast) {                           // prefetch K rows 2 ahead
      __builtin_prefetch(krow0 + 2 * HEAD, 0, 1);
      __builtin_prefetch(krow1 + 2 * HEAD, 0, 1);
    }
    v8f sA0 = {}, sA1 = {}, sB0 = {}, sB1 = {};
    sA0 = wmma_bf16(qa00, kf00, sA0);
    sB0 = wmma_bf16(qa10, kf00, sB0);
    sA0 = wmma_bf16(qa01, kf01, sA0);
    sB0 = wmma_bf16(qa11, kf01, sB0);
    sA1 = wmma_bf16(qa00, kf10, sA1);
    sB1 = wmma_bf16(qa10, kf10, sB1);
    sA1 = wmma_bf16(qa01, kf11, sA1);
    sB1 = wmma_bf16(qa11, kf11, sB1);

    // ---- online softmax + P -> LDS for both q-tiles
    softmax_update(sA0, sA1, qb2,      kb, n, mrow, rmA, rsA,
                   oA[0], oA[1], oA[2], oA[3], lds);
    softmax_update(sB0, sB1, qb2 + 16, kb, n, mrow, rmB, rsB,
                   oB[0], oB[1], oB[2], oB[3], lds2);
    asm volatile("s_wait_dscnt 0x0" ::: "memory");    // order LDS stores -> loads

    // ---- P @ V, V fragments shared between both q-tiles
    v16bf pa0 = load_a_frag(lds  + n * 32, c0a);
    v16bf pa1 = load_a_frag(lds2 + n * 32, c0a);
    const __bf16* vbase = Vh + kb + c0b;
#pragma unroll
    for (int dt = 0; dt < 4; ++dt) {
      v16bf vf = load_b_frag(vbase + (size_t)(dt * 16 + n) * SEQ);
      oA[dt] = wmma_bf16(pa0, vf, oA[dt]);
      oB[dt] = wmma_bf16(pa1, vf, oB[dt]);
    }
    asm volatile("s_wait_dscnt 0x0" ::: "memory");    // guard WAR on P tiles
  }

  // ---- normalize, write y as bf16 in [B*T, 768]
  int b = bh / N_HEAD, h = bh % N_HEAD;
#pragma unroll
  for (int r = 0; r < 8; ++r) {
    float invA = 1.0f / rsA[r];
    float invB = 1.0f / rsB[r];
    size_t rowA = (size_t)b * SEQ + qb2 + r + mrow;
    __bf16* ypA = Y + rowA * N_EMBD + h * HEAD + n;
    __bf16* ypB = ypA + (size_t)16 * N_EMBD;
    ypA[0]  = (__bf16)(oA[0][r] * invA);
    ypA[16] = (__bf16)(oA[1][r] * invA);
    ypA[32] = (__bf16)(oA[2][r] * invA);
    ypA[48] = (__bf16)(oA[3][r] * invA);
    ypB[0]  = (__bf16)(oB[0][r] * invB);
    ypB[16] = (__bf16)(oB[1][r] * invB);
    ypB[32] = (__bf16)(oB[2][r] * invB);
    ypB[48] = (__bf16)(oB[3][r] * invB);
  }
}

// ---------------------------------------------------------------- launch
extern "C" void kernel_launch(void* const* d_in, const int* in_sizes, int n_in,
                              void* d_out, int out_size, void* d_ws, size_t ws_size,
                              hipStream_t stream) {
  const float* x  = (const float*)d_in[0];
  const float* Wk = (const float*)d_in[1];
  const float* Wq = (const float*)d_in[2];
  const float* Wv = (const float*)d_in[3];
  const float* Wp = (const float*)d_in[4];
  const float* bp = (const float*)d_in[5];

  const size_t MTOT = (size_t)BATCH * SEQ;           // 8192 rows
  const size_t XE   = MTOT * N_EMBD;                 // 6291456 elems
  const size_t WE   = (size_t)N_EMBD * N_EMBD;       // 589824 elems

  char* ws = (char*)d_ws;
  size_t off = 0;
  auto alloc = [&](size_t bytes) {
    char* p = ws + off;
    off += (bytes + 255) & ~(size_t)255;
    return p;
  };
  __bf16* xb  = (__bf16*)alloc(XE * 2);
  __bf16* wqb = (__bf16*)alloc(WE * 2);
  __bf16* wkb = (__bf16*)alloc(WE * 2);
  __bf16* wvb = (__bf16*)alloc(WE * 2);
  __bf16* wpb = (__bf16*)alloc(WE * 2);
  __bf16* Qb  = (__bf16*)alloc(XE * 2);              // [B,H,T,64]
  __bf16* Kb  = (__bf16*)alloc(XE * 2);              // [B,H,T,64]
  __bf16* Vtb = (__bf16*)alloc(XE * 2);              // [B,H,64,T]
  __bf16* yb  = (__bf16*)alloc(XE * 2);              // [B*T, 768]

  // 1) fp32 -> bf16 casts
  cast_f32_to_bf16<<<4096, 256, 0, stream>>>(x,  xb,  (int)XE);
  cast_f32_to_bf16<<<1152, 256, 0, stream>>>(Wq, wqb, (int)WE);
  cast_f32_to_bf16<<<1152, 256, 0, stream>>>(Wk, wkb, (int)WE);
  cast_f32_to_bf16<<<1152, 256, 0, stream>>>(Wv, wvb, (int)WE);
  cast_f32_to_bf16<<<1152, 256, 0, stream>>>(Wp, wpb, (int)WE);

  // 2) QKV projections: 512 M-tiles x 12 N64-tiles = 6144 tiles, 8 waves/block
  const int GEMM_BLOCKS = (512 * 12) / 8;            // 768
  gemm_wmma<<<GEMM_BLOCKS, 256, 0, stream>>>(xb, wqb, Qb, nullptr, nullptr, 0);
  gemm_wmma<<<GEMM_BLOCKS, 256, 0, stream>>>(xb, wkb, Kb, nullptr, nullptr, 0);
  gemm_wmma<<<GEMM_BLOCKS, 256, 0, stream>>>(xb, wvb, Vtb, nullptr, nullptr, 1);

  // 3) flash attention: B*H*(T/32) = 3072 supertiles, 8 waves/block
  attn_flash<<<3072 / 8, 256, 0, stream>>>(Qb, Kb, Vtb, yb);

  // 4) output projection + bias -> fp32 d_out
  gemm_wmma<<<GEMM_BLOCKS, 256, 0, stream>>>(yb, wpb, nullptr, (float*)d_out, bp, 2);
}
